// SAINT_70411693850858
// MI455X (gfx1250) — matrix operand order repaired and checked by reading, not compiled, not run on hardware
//
#include <hip/hip_runtime.h>

#define N_NODES 50000
#define N_EDGES 600000
#define NHID    128
#define NCLASS  40
#define KTOT    256   // concatenated K = [agg | x]

typedef __attribute__((ext_vector_type(16))) _Float16 v16h;
typedef __attribute__((ext_vector_type(8)))  _Float16 v8h;
typedef __attribute__((ext_vector_type(8)))  float    v8f;
typedef __attribute__((ext_vector_type(4)))  float    v4f;

// ---------------------------------------------------------------- zero agg
__global__ void zero_f32(float* __restrict__ p, int n4) {
  int i = blockIdx.x * blockDim.x + threadIdx.x;
  if (i < n4) ((v4f*)p)[i] = (v4f){0.f, 0.f, 0.f, 0.f};
}

// ------------------------------------------------- weights f32 -> f16, K-major
// Wt layout: [n][k], n in 0..127, k in 0..255 (k<128 -> Wl[k][n], else Wr[k-128][n])
__global__ void cvt_weights(const float* __restrict__ Wl,
                            const float* __restrict__ Wr,
                            _Float16* __restrict__ Wt) {
  int t = blockIdx.x * blockDim.x + threadIdx.x;   // 128*256 threads
  int n = t >> 8;
  int k = t & 255;
  float v = (k < NHID) ? Wl[k * NHID + n] : Wr[(k - NHID) * NHID + n];
  Wt[t] = (_Float16)v;
}

// -------------------------------------------- edge gather * w -> atomic scatter
// 32 lanes per edge, float4 per lane (128 feats). Source & dest are L2-resident.
__global__ void sage_scatter(const float* __restrict__ xsrc,
                             const long long* __restrict__ ei,
                             const float* __restrict__ ew,
                             float* __restrict__ agg) {
  int e    = blockIdx.x * 8 + (threadIdx.x >> 5);
  int lane = threadIdx.x & 31;
  if (e >= N_EDGES) return;
  int   row = (int)ei[e];             // destination
  int   col = (int)ei[N_EDGES + e];   // source
  float w   = ew[e];
  v4f v = ((const v4f*)(xsrc + (size_t)col * NHID))[lane];
  float* dst = agg + (size_t)row * NHID + lane * 4;
  atomicAdd(dst + 0, v[0] * w);
  atomicAdd(dst + 1, v[1] * w);
  atomicAdd(dst + 2, v[2] * w);
  atomicAdd(dst + 3, v[3] * w);
}

// ---------------------------------- out = relu([agg|x] @ Wt^T(f16) + bias)
// block = 256 thr = 8 waves; block owns 16 rows; wave w owns cols [16w,16w+16)
__launch_bounds__(256)
__global__ void sage_gemm_wmma(const float* __restrict__ A0,     // agg  [N,128]
                               const float* __restrict__ A1,     // x    [N,128]
                               const _Float16* __restrict__ Wt,  // [128][256] n-major
                               const float* __restrict__ bias,   // [128]
                               float* __restrict__ out) {        // [N,128]
  __shared__ _Float16 As[16][KTOT + 8];            // 16 x 256 f16 tile (+pad)
  const int tile = blockIdx.x;                     // 0..3124 (N = 3125*16)
  const int tid  = threadIdx.x;

  // cooperative A-tile load, f32 -> f16
  #pragma unroll
  for (int s = 0; s < 16; ++s) {
    int i    = tid + (s << 8);
    int row  = i >> 8;
    int colk = i & 255;
    float v = (colk < NHID)
        ? A0[(size_t)(tile * 16 + row) * NHID + colk]
        : A1[(size_t)(tile * 16 + row) * NHID + (colk - NHID)];
    As[row][colk] = (_Float16)v;
  }
  __syncthreads();

  const int wave = tid >> 5;
  const int lane = tid & 31;
  const int mA   = lane & 15;            // A row
  const int koA  = (lane >> 4) * 8;      // A half-wave K select
  const int nB   = (wave << 4) + (lane & 15);  // global output column
  const int kbB  = (lane >> 4) * 8;      // B half-wave K select

  v8f acc = {};
  #pragma unroll
  for (int kk = 0; kk < KTOT; kk += 32) {
    // A frag: lane<16 -> K{kk+0..7, kk+16..23}; lane>=16 -> K{kk+8..15, kk+24..31}
    v8h alo = *(const v8h*)&As[mA][kk + koA];
    v8h ahi = *(const v8h*)&As[mA][kk + 16 + koA];
    // B frag: lane<16 -> K{kk+0..7, kk+16..23}; lane>=16 -> K{kk+8..15, kk+24..31}
    const _Float16* bp = Wt + (size_t)nB * KTOT + kk + kbB;
    v8h blo = *(const v8h*)bp;
    v8h bhi = *(const v8h*)(bp + 16);
    v16h a, b;
    #pragma unroll
    for (int i = 0; i < 8; ++i) {
      a[i] = alo[i]; a[8 + i] = ahi[i];
      b[i] = blo[i]; b[8 + i] = bhi[i];
    }
    acc = __builtin_amdgcn_wmma_f32_16x16x32_f16(
        /*neg_a=*/false, a, /*neg_b=*/false, b,
        /*c_mod=*/(short)0, acc, /*reuse_a=*/false, /*reuse_b=*/false);
  }

  // epilogue: D VGPR j -> M = j + 8*(lane>=16), N = lane&15
  const float bn    = bias[nB];
  const int   mbase = tile * 16 + ((lane >> 4) << 3);
  #pragma unroll
  for (int j = 0; j < 8; ++j) {
    float v = acc[j] + bn;
    out[(size_t)(mbase + j) * NHID + nB] = v > 0.f ? v : 0.f;
  }
}

// ------------------- logits = [x1|x2|x3] @ Wlin + blin; row log_softmax (40)
// one wave32 per node; lane L owns class L, lanes 0..7 also own class 32+L
__launch_bounds__(256)
__global__ void head_logsoftmax(const float* __restrict__ x1,
                                const float* __restrict__ x2,
                                const float* __restrict__ x3,
                                const float* __restrict__ Wlin,  // [384][40]
                                const float* __restrict__ blin,  // [40]
                                float* __restrict__ out) {       // [N][40]
  int r    = blockIdx.x * 8 + (threadIdx.x >> 5);
  int lane = threadIdx.x & 31;
  if (r >= N_NODES) return;

  float acc0 = blin[lane];
  float acc1 = (lane < 8) ? blin[32 + lane] : 0.f;

  const float* rows[3] = { x1 + (size_t)r * NHID,
                           x2 + (size_t)r * NHID,
                           x3 + (size_t)r * NHID };
  for (int part = 0; part < 3; ++part) {
    const float* hr = rows[part];
    for (int k0 = 0; k0 < NHID; k0 += 32) {
      float hv = hr[k0 + lane];
      #pragma unroll
      for (int j = 0; j < 32; ++j) {
        float v = __shfl(hv, j, 32);
        const float* wrow = Wlin + (size_t)(part * NHID + k0 + j) * NCLASS;
        acc0 += v * wrow[lane];
        if (lane < 8) acc1 += v * wrow[32 + lane];
      }
    }
  }

  // log_softmax over 40 classes (wave32 reductions)
  float m = fmaxf(acc0, (lane < 8) ? acc1 : -3.4e38f);
  #pragma unroll
  for (int o = 16; o > 0; o >>= 1) m = fmaxf(m, __shfl_xor(m, o, 32));
  float s = __expf(acc0 - m) + ((lane < 8) ? __expf(acc1 - m) : 0.f);
  #pragma unroll
  for (int o = 16; o > 0; o >>= 1) s += __shfl_xor(s, o, 32);
  float lz = m + __logf(s);

  out[(size_t)r * NCLASS + lane] = acc0 - lz;
  if (lane < 8) out[(size_t)r * NCLASS + 32 + lane] = acc1 - lz;
}

extern "C" void kernel_launch(void* const* d_in, const int* in_sizes, int n_in,
                              void* d_out, int out_size, void* d_ws, size_t ws_size,
                              hipStream_t stream) {
  const float*     x  = (const float*)d_in[0];
  const long long* ei = (const long long*)d_in[1];   // edge_index int64 [2,E]
  const float*     ew = (const float*)d_in[2];
  const float* Wl[3] = {(const float*)d_in[3], (const float*)d_in[6], (const float*)d_in[9]};
  const float* Wr[3] = {(const float*)d_in[4], (const float*)d_in[7], (const float*)d_in[10]};
  const float* bb[3] = {(const float*)d_in[5], (const float*)d_in[8], (const float*)d_in[11]};
  const float* Wlin = (const float*)d_in[12];
  const float* blin = (const float*)d_in[13];

  const size_t NM = (size_t)N_NODES * NHID;
  float* agg = (float*)d_ws;
  float* xs0 = agg + NM;
  float* xs1 = xs0 + NM;
  float* xs2 = xs1 + NM;
  float* xs[3] = { xs0, xs1, xs2 };
  _Float16* Wt0 = (_Float16*)(xs2 + NM);
  _Float16* Wt[3] = { Wt0, Wt0 + KTOT * NHID, Wt0 + 2 * KTOT * NHID };

  for (int l = 0; l < 3; ++l)
    cvt_weights<<<(KTOT * NHID) / 256, 256, 0, stream>>>(Wl[l], Wr[l], Wt[l]);

  const float* cur = x;
  for (int l = 0; l < 3; ++l) {
    zero_f32<<<(int)((NM / 4 + 255) / 256), 256, 0, stream>>>(agg, (int)(NM / 4));
    sage_scatter<<<(N_EDGES + 7) / 8, 256, 0, stream>>>(cur, ei, ew, agg);
    sage_gemm_wmma<<<N_NODES / 16, 256, 0, stream>>>(agg, cur, Wt[l], bb[l], xs[l]);
    cur = xs[l];
  }

  head_logsoftmax<<<(N_NODES + 7) / 8, 256, 0, stream>>>(
      xs[0], xs[1], xs[2], Wlin, blin, (float*)d_out);
}
